// TemporalMultiHeadAttention_48627619725849
// MI455X (gfx1250) — matrix operand order repaired and checked by reading, not compiled, not run on hardware
//
#include <hip/hip_runtime.h>
#include <hip/hip_bf16.h>

#define EMBED 1024
#define NHEADS 16
#define HD 64
#define BATCH 2
#define SEQ 2048
#define MTOT (BATCH*SEQ)   // 4096

typedef __bf16 bf16_t;
typedef __bf16 v16bf __attribute__((ext_vector_type(16)));
typedef float  v8f   __attribute__((ext_vector_type(8)));
typedef unsigned int u32x4 __attribute__((ext_vector_type(4)));

union FragU { v16bf v; u32x4 q[2]; };

// ---- WMMA fragment loaders (CDNA5 ISA 7.12.2 layouts, wave32) ----

// A-fragment: 16(M) x 32(K) bf16 from row-major [m][k] tile (base at tile origin)
__device__ __forceinline__ v16bf load_a_frag(const bf16_t* base, int ld, int lane) {
  int m  = lane & 15;
  int kb = (lane >> 4) << 3;            // lanes 0-15: k 0..7/16..23, lanes 16-31: k 8..15/24..31
  const bf16_t* p = base + m * ld + kb;
  FragU f;
  f.q[0] = *(const u32x4*)(p);          // 8 bf16 (16B)
  f.q[1] = *(const u32x4*)(p + 16);     // 8 bf16 (16B)
  return f.v;
}

// B-fragment: 32(K) x 16(N) bf16, source stored row-major [n][k]
__device__ __forceinline__ v16bf load_b_frag(const bf16_t* base, int ld, int lane) {
  int n  = lane & 15;
  int kb = (lane >> 4) << 3;            // lanes 0-15: k 0..7/16..23, lanes 16-31: k 8..15/24..31
  const bf16_t* p = base + n * ld + kb;
  FragU f;
  f.q[0] = *(const u32x4*)(p);
  f.q[1] = *(const u32x4*)(p + 16);
  return f.v;
}

__device__ __forceinline__ v8f wmma_bf16(v16bf a, v16bf b, v8f c) {
  return __builtin_amdgcn_wmma_f32_16x16x32_bf16(false, a, false, b, (short)0, c, false, false);
}

__device__ __forceinline__ float redmax16(float v) {
  #pragma unroll
  for (int m = 1; m < 16; m <<= 1) v = fmaxf(v, __shfl_xor(v, m, 32));
  return v;
}
__device__ __forceinline__ float redsum16(float v) {
  #pragma unroll
  for (int m = 1; m < 16; m <<= 1) v += __shfl_xor(v, m, 32);
  return v;
}

// ---------------- fp32 -> bf16 cast ----------------
__global__ void cast_f32_bf16(const float* __restrict__ src, bf16_t* __restrict__ dst, int n) {
  for (int i = blockIdx.x * blockDim.x + threadIdx.x; i < n; i += gridDim.x * blockDim.x)
    dst[i] = (bf16_t)src[i];
}

// ---------------- shared 32x64 wave-tile GEMM core (K = EMBED) ----------------
// Software-pipelined: next K-step fragments are loaded before current WMMAs issue,
// so VMEM latency hides behind the 8-WMMA burst. pA at (m0,0) ld=EMBED,
// pB row-major [n][k] at (n0,0) ld=EMBED. Accumulators: c[mt*4+nt], mt=0..1, nt=0..3.
__device__ __forceinline__ void gemm_tile_32x64(const bf16_t* __restrict__ pA,
                                                const bf16_t* __restrict__ pB,
                                                int lane, v8f c[8]) {
  v16bf a0 = load_a_frag(pA, EMBED, lane);
  v16bf a1 = load_a_frag(pA + 16 * EMBED, EMBED, lane);
  v16bf b0 = load_b_frag(pB +  0 * EMBED, EMBED, lane);
  v16bf b1 = load_b_frag(pB + 16 * EMBED, EMBED, lane);
  v16bf b2 = load_b_frag(pB + 32 * EMBED, EMBED, lane);
  v16bf b3 = load_b_frag(pB + 48 * EMBED, EMBED, lane);

  for (int k0 = 0; k0 < EMBED; k0 += 32) {
    int kn = (k0 + 32 < EMBED) ? k0 + 32 : k0;   // clamp: last prefetch is redundant, not OOB
    if (k0 + 64 < EMBED) {                        // L2 prefetch two steps ahead (global_prefetch_b8)
      __builtin_prefetch(pA + k0 + 64, 0, 0);
      __builtin_prefetch(pB + k0 + 64, 0, 0);
    }
    v16bf na0 = load_a_frag(pA + kn, EMBED, lane);
    v16bf na1 = load_a_frag(pA + 16 * EMBED + kn, EMBED, lane);
    v16bf nb0 = load_b_frag(pB +  0 * EMBED + kn, EMBED, lane);
    v16bf nb1 = load_b_frag(pB + 16 * EMBED + kn, EMBED, lane);
    v16bf nb2 = load_b_frag(pB + 32 * EMBED + kn, EMBED, lane);
    v16bf nb3 = load_b_frag(pB + 48 * EMBED + kn, EMBED, lane);

    c[0] = wmma_bf16(a0, b0, c[0]);
    c[1] = wmma_bf16(a0, b1, c[1]);
    c[2] = wmma_bf16(a0, b2, c[2]);
    c[3] = wmma_bf16(a0, b3, c[3]);
    c[4] = wmma_bf16(a1, b0, c[4]);
    c[5] = wmma_bf16(a1, b1, c[5]);
    c[6] = wmma_bf16(a1, b2, c[6]);
    c[7] = wmma_bf16(a1, b3, c[7]);

    a0 = na0; a1 = na1; b0 = nb0; b1 = nb1; b2 = nb2; b3 = nb3;
  }
}

// ---------------- QKV projection GEMM ----------------
// Block: 8 waves as 2(M) x 4(N) -> 64 x 256 block tile. Scatter into Q/K/Vt bf16.
__global__ __launch_bounds__(256) void qkv_gemm(const bf16_t* __restrict__ Xb,
                                               const bf16_t* __restrict__ Wb,
                                               const float*  __restrict__ bias,
                                               bf16_t* __restrict__ Q,
                                               bf16_t* __restrict__ K,
                                               bf16_t* __restrict__ Vt) {
  const int lane = threadIdx.x & 31;
  const int wave = threadIdx.x >> 5;
  const int m0 = blockIdx.x * 64  + (wave & 1) * 32;
  const int n0 = blockIdx.y * 256 + (wave >> 1) * 64;

  v8f c[8];
  #pragma unroll
  for (int i = 0; i < 8; i++) c[i] = (v8f){0.f,0.f,0.f,0.f,0.f,0.f,0.f,0.f};

  gemm_tile_32x64(Xb + (size_t)m0 * EMBED, Wb + (size_t)n0 * EMBED, lane, c);

  #pragma unroll
  for (int mt = 0; mt < 2; mt++) {
    const int mbase = m0 + mt * 16 + ((lane >> 4) << 3);
    #pragma unroll
    for (int nt = 0; nt < 4; nt++) {
      int n = n0 + nt * 16 + (lane & 15);
      float bv = bias[n];
      int h = n / 192, j = n % 192;
      int sel = j >> 6, d = j & 63;
      #pragma unroll
      for (int r = 0; r < 8; r++) {
        int m = mbase + r;
        int b_ = m >> 11, s = m & (SEQ - 1);
        bf16_t o = (bf16_t)(c[mt * 4 + nt][r] + bv);
        int hb = b_ * NHEADS + h;
        if (sel == 0)      Q [((size_t)hb * SEQ + s) * HD + d] = o;
        else if (sel == 1) K [((size_t)hb * SEQ + s) * HD + d] = o;
        else               Vt[((size_t)hb * HD  + d) * SEQ + s] = o;
      }
    }
  }
}

// ---------------- flash attention: softmax(Q K^T / 8) @ V ----------------
// One wave per 16-row Q tile. K fragments double-buffered across t-chunks and V
// fragments issued before the softmax VALU burst so loads hide behind v_exp work.
__global__ __launch_bounds__(256) void attention_kernel(const bf16_t* __restrict__ Q,
                                                        const bf16_t* __restrict__ K,
                                                        const bf16_t* __restrict__ Vt,
                                                        bf16_t* __restrict__ attnb) {
  __shared__ __align__(64) bf16_t Plds[8][16][32];   // 8 KB, per-wave private tiles

  const int lane = threadIdx.x & 31;
  const int wave = threadIdx.x >> 5;
  const int bh = blockIdx.y;            // b*H + h
  const int b  = bh >> 4, h = bh & 15;
  const int q0 = blockIdx.x * 128 + wave * 16;

  const bf16_t* Qh = Q  + (size_t)bh * SEQ * HD;
  const bf16_t* Kh = K  + (size_t)bh * SEQ * HD;
  const bf16_t* Vh = Vt + (size_t)bh * HD * SEQ;

  v16bf qa0 = load_a_frag(Qh + (size_t)q0 * HD +  0, HD, lane);
  v16bf qa1 = load_a_frag(Qh + (size_t)q0 * HD + 32, HD, lane);

  v8f o[4];
  #pragma unroll
  for (int i = 0; i < 4; i++) o[i] = (v8f){0.f,0.f,0.f,0.f,0.f,0.f,0.f,0.f};
  float mrow[8], lrow[8];
  #pragma unroll
  for (int r = 0; r < 8; r++) { mrow[r] = -1e30f; lrow[r] = 0.f; }

  // K fragments for chunk 0 (double-buffered across the t-loop)
  v16bf kb0 = load_b_frag(Kh +  0 * HD +  0, HD, lane);
  v16bf kb1 = load_b_frag(Kh +  0 * HD + 32, HD, lane);
  v16bf kb2 = load_b_frag(Kh + 16 * HD +  0, HD, lane);
  v16bf kb3 = load_b_frag(Kh + 16 * HD + 32, HD, lane);

  for (int t0 = 0; t0 < SEQ; t0 += 32) {
    // V fragments for this chunk: issue now, consumed only after softmax
    v16bf vb0 = load_b_frag(Vh + (size_t)( 0) * SEQ + t0, SEQ, lane);
    v16bf vb1 = load_b_frag(Vh + (size_t)(16) * SEQ + t0, SEQ, lane);
    v16bf vb2 = load_b_frag(Vh + (size_t)(32) * SEQ + t0, SEQ, lane);
    v16bf vb3 = load_b_frag(Vh + (size_t)(48) * SEQ + t0, SEQ, lane);

    v8f s0 = (v8f){0.f,0.f,0.f,0.f,0.f,0.f,0.f,0.f};
    v8f s1 = (v8f){0.f,0.f,0.f,0.f,0.f,0.f,0.f,0.f};
    s0 = wmma_bf16(qa0, kb0, s0);
    s0 = wmma_bf16(qa1, kb1, s0);
    s1 = wmma_bf16(qa0, kb2, s1);
    s1 = wmma_bf16(qa1, kb3, s1);

    // prefetch next chunk's K fragments (redundant reload on last iteration)
    int tn = (t0 + 32 < SEQ) ? t0 + 32 : t0;
    kb0 = load_b_frag(Kh + (size_t)(tn +  0) * HD +  0, HD, lane);
    kb1 = load_b_frag(Kh + (size_t)(tn +  0) * HD + 32, HD, lane);
    kb2 = load_b_frag(Kh + (size_t)(tn + 16) * HD +  0, HD, lane);
    kb3 = load_b_frag(Kh + (size_t)(tn + 16) * HD + 32, HD, lane);

    // online softmax over the 32 columns (VALU work overlaps in-flight loads)
    #pragma unroll
    for (int r = 0; r < 8; r++) {
      float a = s0[r] * 0.125f;           // 1/sqrt(64)
      float cc = s1[r] * 0.125f;
      float rm = redmax16(fmaxf(a, cc));
      float mnew = fmaxf(mrow[r], rm);
      float corr = __expf(mrow[r] - mnew);
      mrow[r] = mnew;
      float p0 = __expf(a - mnew);
      float p1 = __expf(cc - mnew);
      lrow[r] = lrow[r] * corr + redsum16(p0 + p1);
      #pragma unroll
      for (int dt = 0; dt < 4; dt++) o[dt][r] *= corr;
      int row = r + ((lane >> 4) << 3);
      Plds[wave][row][lane & 15]        = (bf16_t)p0;
      Plds[wave][row][16 + (lane & 15)] = (bf16_t)p1;
    }

    // C-fragment -> A-fragment conversion via per-wave LDS tile (DS in-order in-wave)
    v16bf pa = load_a_frag(&Plds[wave][0][0], 32, lane);
    o[0] = wmma_bf16(pa, vb0, o[0]);
    o[1] = wmma_bf16(pa, vb1, o[1]);
    o[2] = wmma_bf16(pa, vb2, o[2]);
    o[3] = wmma_bf16(pa, vb3, o[3]);
  }

  // Epilogue: normalize and scatter through the reference's (B,hd,H,S)->(B,S,E) view.
  #pragma unroll
  for (int r = 0; r < 8; r++) {
    float inv = 1.0f / lrow[r];
    int srow = q0 + r + ((lane >> 4) << 3);
    #pragma unroll
    for (int dt = 0; dt < 4; dt++) {
      int d = dt * 16 + (lane & 15);
      int flat = d * (NHEADS * SEQ) + h * SEQ + srow;      // index inside batch b
      attnb[(size_t)b * (SEQ * EMBED) + flat] = (bf16_t)(o[dt][r] * inv);
    }
  }
}

// ---------------- output projection GEMM (fp32 out) ----------------
__global__ __launch_bounds__(256) void out_gemm(const bf16_t* __restrict__ Ab,
                                                const bf16_t* __restrict__ Wb,
                                                const float*  __restrict__ bias,
                                                float* __restrict__ out) {
  const int lane = threadIdx.x & 31;
  const int wave = threadIdx.x >> 5;
  const int m0 = blockIdx.x * 64  + (wave & 1) * 32;
  const int n0 = blockIdx.y * 256 + (wave >> 1) * 64;

  v8f c[8];
  #pragma unroll
  for (int i = 0; i < 8; i++) c[i] = (v8f){0.f,0.f,0.f,0.f,0.f,0.f,0.f,0.f};

  gemm_tile_32x64(Ab + (size_t)m0 * EMBED, Wb + (size_t)n0 * EMBED, lane, c);

  #pragma unroll
  for (int mt = 0; mt < 2; mt++) {
    const int mbase = m0 + mt * 16 + ((lane >> 4) << 3);
    #pragma unroll
    for (int nt = 0; nt < 4; nt++) {
      int n = n0 + nt * 16 + (lane & 15);
      float bv = bias[n];
      #pragma unroll
      for (int r = 0; r < 8; r++)
        out[(size_t)(mbase + r) * EMBED + n] = c[mt * 4 + nt][r] + bv;
    }
  }
}

extern "C" void kernel_launch(void* const* d_in, const int* in_sizes, int n_in,
                              void* d_out, int out_size, void* d_ws, size_t ws_size,
                              hipStream_t stream) {
  const float* x     = (const float*)d_in[0];
  const float* qkv_w = (const float*)d_in[1];
  const float* qkv_b = (const float*)d_in[2];
  const float* out_w = (const float*)d_in[3];
  const float* out_b = (const float*)d_in[4];
  float* out = (float*)d_out;

  char* ws = (char*)d_ws;
  size_t off = 0;
  auto alloc = [&](size_t bytes) -> void* {
    void* p = ws + off;
    off += (bytes + 255) & ~(size_t)255;
    return p;
  };

  bf16_t* Xb    = (bf16_t*)alloc((size_t)MTOT * EMBED * 2);               // 8 MB
  bf16_t* Wqkvb = (bf16_t*)alloc((size_t)3 * EMBED * EMBED * 2);          // 6 MB
  bf16_t* Woutb = (bf16_t*)alloc((size_t)EMBED * EMBED * 2);              // 2 MB
  bf16_t* Qb    = (bf16_t*)alloc((size_t)BATCH * NHEADS * SEQ * HD * 2);  // 8 MB
  bf16_t* Kb    = (bf16_t*)alloc((size_t)BATCH * NHEADS * SEQ * HD * 2);  // 8 MB
  bf16_t* Vtb   = (bf16_t*)alloc((size_t)BATCH * NHEADS * HD * SEQ * 2);  // 8 MB
  bf16_t* Attnb = (bf16_t*)alloc((size_t)MTOT * EMBED * 2);               // 8 MB

  cast_f32_bf16<<<1024, 256, 0, stream>>>(x,     Xb,    MTOT * EMBED);
  cast_f32_bf16<<<1024, 256, 0, stream>>>(qkv_w, Wqkvb, 3 * EMBED * EMBED);
  cast_f32_bf16<<<1024, 256, 0, stream>>>(out_w, Woutb, EMBED * EMBED);

  qkv_gemm<<<dim3(MTOT / 64, (3 * EMBED) / 256), 256, 0, stream>>>(
      Xb, Wqkvb, qkv_b, Qb, Kb, Vtb);

  attention_kernel<<<dim3(SEQ / 128, BATCH * NHEADS), 256, 0, stream>>>(
      Qb, Kb, Vtb, Attnb);

  out_gemm<<<dim3(MTOT / 64, EMBED / 256), 256, 0, stream>>>(
      Attnb, Woutb, out_b, out);
}
